// LSTMCreature_7567732375853
// MI455X (gfx1250) — matrix operand order repaired and checked by reading, not compiled, not run on hardware
//
#include <hip/hip_runtime.h>
#include <hip/hip_bf16.h>

typedef __attribute__((ext_vector_type(16))) _Float16 v16h;
typedef __attribute__((ext_vector_type(8)))  _Float16 v8h;
typedef __attribute__((ext_vector_type(8)))  float    v8f;

#define LSTM_B   16384
#define LSTM_IN  128
#define LSTM_H   256
#define LSTM_OUT 64

// ---------------------------------------------------------------- helpers
__device__ __forceinline__ float fsigmoid(float x) {
    return 1.0f / (1.0f + __expf(-x));
}
__device__ __forceinline__ float ftanh(float x) {
    x = fminf(fmaxf(x, -15.0f), 15.0f);
    float e = __expf(2.0f * x);
    return (e - 1.0f) / (e + 1.0f);
}

__device__ __forceinline__ v16h load_a_frag(const _Float16* arow, int kofsA, int kk) {
    // A layout (16-bit 16x32): lane = row; lanes<16 hold K{0..7,16..23},
    // lanes>=16 hold K{8..15,24..31}.
    v8h a0 = *(const v8h*)(arow + kk + kofsA);
    v8h a1 = *(const v8h*)(arow + kk + kofsA + 16);
    v16h a;
#pragma unroll
    for (int i = 0; i < 8; ++i) { a[i] = a0[i]; a[8 + i] = a1[i]; }
    return a;
}

// ------------------------------------------------------------- converters
__global__ void cvt_f32_to_f16_kernel(const float* __restrict__ src,
                                      _Float16* __restrict__ dst, int n) {
    int i = blockIdx.x * blockDim.x + threadIdx.x;
    if (i < n) dst[i] = (_Float16)src[i];
}

__global__ void bias_sum_kernel(const float* __restrict__ a,
                                const float* __restrict__ b,
                                float* __restrict__ o, int n) {
    int i = blockIdx.x * blockDim.x + threadIdx.x;
    if (i < n) o[i] = a[i] + b[i];
}

// ---------------------------------------------------- fused LSTM layer
// One timestep, h0=c0=0 =>  h = sigmoid(o) * tanh( sigmoid(i) * tanh(g) )
// Gates i,g,o are rows [0:256), [512:768), [768:1024) of W_ih (f is dead).
// Block: 3 waves (96 thr). Wave w owns gate w and computes 64 rows x 16 cols:
// one B (weight) fragment per k-step is reused across 4 row-tile WMMAs.
template <int K>
__global__ __launch_bounds__(96) void lstm_layer_kernel(
    const _Float16* __restrict__ A,    // [B, K] f16 row-major
    const _Float16* __restrict__ W,    // [1024, K] f16 row-major
    const float* __restrict__ bias,    // [1024] = b_ih + b_hh
    _Float16* __restrict__ H)          // [B, 256] f16
{
    __shared__ float gtile[3 * 4 * 256];           // [gate][rowTile][m][j]

    const int lane = threadIdx.x & 31;
    const int wave = threadIdx.x >> 5;             // 0=i, 1=g, 2=o
    const int row0 = blockIdx.x * 64;              // batch tile (4 x 16 rows)
    const int col0 = blockIdx.y * 16;              // hidden tile
    const int gateBase = (wave == 0) ? 0 : ((wave == 1) ? 512 : 768);

    const int kofsA = (lane & 16) ? 8 : 0;
    const _Float16* arow0 = A + (size_t)(row0 + (lane & 15)) * K;
    // B layout (32x16): lane = column; lanes<16 hold K{0..7,16..23}, lanes>=16 K{8..15,24..31}.
    const _Float16* brow = W + (size_t)(gateBase + col0 + (lane & 15)) * K
                             + ((lane & 16) ? 8 : 0);

    v8f acc[4] = {{}, {}, {}, {}};
    for (int kk = 0; kk < K; kk += 32) {
        v16h b = load_a_frag(brow, 0, kk);
#pragma unroll
        for (int t = 0; t < 4; ++t) {
            v16h a = load_a_frag(arow0 + (size_t)(t * 16) * K, kofsA, kk);
            acc[t] = __builtin_amdgcn_wmma_f32_16x16x32_f16(
                         false, a, false, b, (short)0, acc[t], false, false);
        }
    }

    // C/D layout: lane = column n (mod 16); rows m = vgpr(+8 for lanes>=16).
    {
        const int n = lane & 15;
        const int mbase = (lane & 16) ? 8 : 0;
#pragma unroll
        for (int t = 0; t < 4; ++t)
#pragma unroll
            for (int r = 0; r < 8; ++r)
                gtile[wave * 1024 + t * 256 + (mbase + r) * 16 + n] = acc[t][r];
    }
    __syncthreads();

    const float bi = 0.0f; (void)bi;
    for (int idx = threadIdx.x; idx < 1024; idx += 96) {
        const int j = idx & 15;                    // hidden col within tile
        const int m = idx >> 4;                    // 0..63 within row block
        float iv = gtile[idx]        + bias[      col0 + j];
        float gv = gtile[1024 + idx] + bias[512 + col0 + j];
        float ov = gtile[2048 + idx] + bias[768 + col0 + j];
        float h = fsigmoid(ov) * ftanh(fsigmoid(iv) * ftanh(gv));
        H[(size_t)(row0 + m) * LSTM_H + col0 + j] = (_Float16)h;
    }
}

// ------------------------------------------- FC head + tanh/softmax mixing
// out = H2 @ fc_W.T + fc_b;  w = tanh(out)*softmax(|out|); w /= max(sum|w|,eps)
// Block: 4 waves (128 thr). Wave w owns columns [16w,16w+16) for 64 rows
// (4 row-tile WMMAs reuse one weight fragment). Then 64 threads do the
// per-row softmax head.
__global__ __launch_bounds__(128) void fc_head_kernel(
    const _Float16* __restrict__ A,    // [B, 256] f16
    const _Float16* __restrict__ Wf,   // [64, 256] f16
    const float* __restrict__ bias,    // [64] f32
    float* __restrict__ out)           // [B, 64] f32
{
    __shared__ float tile[64 * 64];
    __shared__ float tsh[64 * 64];

    const int lane = threadIdx.x & 31;
    const int wave = threadIdx.x >> 5;             // 0..3 -> column tile
    const int row0 = blockIdx.x * 64;
    const int n0 = wave * 16;

    const int kofsA = (lane & 16) ? 8 : 0;
    const _Float16* arow0 = A + (size_t)(row0 + (lane & 15)) * LSTM_H;
    const _Float16* brow = Wf + (size_t)(n0 + (lane & 15)) * LSTM_H
                              + ((lane & 16) ? 8 : 0);

    v8f acc[4] = {{}, {}, {}, {}};
    for (int kk = 0; kk < LSTM_H; kk += 32) {
        v16h b = load_a_frag(brow, 0, kk);
#pragma unroll
        for (int t = 0; t < 4; ++t) {
            v16h a = load_a_frag(arow0 + (size_t)(t * 16) * LSTM_H, kofsA, kk);
            acc[t] = __builtin_amdgcn_wmma_f32_16x16x32_f16(
                         false, a, false, b, (short)0, acc[t], false, false);
        }
    }
    {
        const int n = lane & 15;
        const int mbase = (lane & 16) ? 8 : 0;
#pragma unroll
        for (int t = 0; t < 4; ++t)
#pragma unroll
            for (int r = 0; r < 8; ++r)
                tile[(t * 16 + mbase + r) * 64 + n0 + n] = acc[t][r];
    }
    __syncthreads();

    if (threadIdx.x < 64) {
        const int m = threadIdx.x;
        float* v = &tile[m * 64];
        float* t = &tsh[m * 64];
        float amax = -1e30f;
        for (int j = 0; j < 64; ++j) {
            float x = v[j] + bias[j];
            t[j] = ftanh(x);
            float ax = fabsf(x);
            v[j] = ax;
            amax = fmaxf(amax, ax);
        }
        float S = 0.0f;
        for (int j = 0; j < 64; ++j) {
            float e = __expf(v[j] - amax);
            v[j] = e;
            S += e;
        }
        const float invS = 1.0f / S;
        float den = 0.0f;
        for (int j = 0; j < 64; ++j) {
            float w = t[j] * v[j] * invS;
            v[j] = w;
            den += fabsf(w);
        }
        const float invD = 1.0f / fmaxf(den, 1e-12f);
        float* orow = out + (size_t)(row0 + m) * LSTM_OUT;
        for (int j = 0; j < 64; ++j) orow[j] = v[j] * invD;
    }
}

// ----------------------------------------------------------------- launch
extern "C" void kernel_launch(void* const* d_in, const int* in_sizes, int n_in,
                              void* d_out, int out_size, void* d_ws, size_t ws_size,
                              hipStream_t stream) {
    (void)in_sizes; (void)n_in; (void)out_size; (void)ws_size;

    // input order: x, [W_ih, W_hh, b_ih, b_hh] x3, fc_W, fc_b
    const float* x     = (const float*)d_in[0];
    const float* Wih0  = (const float*)d_in[1];
    const float* bih0  = (const float*)d_in[3];
    const float* bhh0  = (const float*)d_in[4];
    const float* Wih1  = (const float*)d_in[5];
    const float* bih1  = (const float*)d_in[7];
    const float* bhh1  = (const float*)d_in[8];
    const float* Wih2  = (const float*)d_in[9];
    const float* bih2  = (const float*)d_in[11];
    const float* bhh2  = (const float*)d_in[12];
    const float* fcW   = (const float*)d_in[13];
    const float* fcb   = (const float*)d_in[14];
    float* out = (float*)d_out;

    // workspace carve-up (all offsets 256B aligned)
    char* ws = (char*)d_ws;
    size_t off = 0;
    auto take = [&](size_t bytes) { char* p = ws + off; off += (bytes + 255) & ~(size_t)255; return p; };
    _Float16* x16   = (_Float16*)take((size_t)LSTM_B * LSTM_IN * 2);   // 4 MB
    _Float16* h0    = (_Float16*)take((size_t)LSTM_B * LSTM_H  * 2);   // 8 MB
    _Float16* h1    = (_Float16*)take((size_t)LSTM_B * LSTM_H  * 2);   // 8 MB
    _Float16* h2    = (_Float16*)take((size_t)LSTM_B * LSTM_H  * 2);   // 8 MB
    _Float16* W016  = (_Float16*)take((size_t)4 * LSTM_H * LSTM_IN * 2);
    _Float16* W116  = (_Float16*)take((size_t)4 * LSTM_H * LSTM_H  * 2);
    _Float16* W216  = (_Float16*)take((size_t)4 * LSTM_H * LSTM_H  * 2);
    _Float16* fW16  = (_Float16*)take((size_t)LSTM_OUT * LSTM_H * 2);
    float*    bias0 = (float*)take((size_t)4 * LSTM_H * 4);
    float*    bias1 = (float*)take((size_t)4 * LSTM_H * 4);
    float*    bias2 = (float*)take((size_t)4 * LSTM_H * 4);

    auto cvt = [&](const float* s, _Float16* d, int n) {
        cvt_f32_to_f16_kernel<<<(n + 255) / 256, 256, 0, stream>>>(s, d, n);
    };
    cvt(x,    x16,  LSTM_B * LSTM_IN);
    cvt(Wih0, W016, 4 * LSTM_H * LSTM_IN);
    cvt(Wih1, W116, 4 * LSTM_H * LSTM_H);
    cvt(Wih2, W216, 4 * LSTM_H * LSTM_H);
    cvt(fcW,  fW16, LSTM_OUT * LSTM_H);
    bias_sum_kernel<<<4, 256, 0, stream>>>(bih0, bhh0, bias0, 4 * LSTM_H);
    bias_sum_kernel<<<4, 256, 0, stream>>>(bih1, bhh1, bias1, 4 * LSTM_H);
    bias_sum_kernel<<<4, 256, 0, stream>>>(bih2, bhh2, bias2, 4 * LSTM_H);

    dim3 lgrid(LSTM_B / 64, LSTM_H / 16);
    lstm_layer_kernel<LSTM_IN><<<lgrid, 96, 0, stream>>>(x16, W016, bias0, h0);
    lstm_layer_kernel<LSTM_H> <<<lgrid, 96, 0, stream>>>(h0,  W116, bias1, h1);
    lstm_layer_kernel<LSTM_H> <<<lgrid, 96, 0, stream>>>(h1,  W216, bias2, h2);

    fc_head_kernel<<<LSTM_B / 64, 128, 0, stream>>>(h2, fW16, fcb, out);
}
